// EHRmamba_50190987821419
// MI455X (gfx1250) — matrix operand order repaired and checked by reading, not compiled, not run on hardware
//
#include <hip/hip_runtime.h>
#include <hip/hip_bf16.h>
#include <math.h>

// ---------------- problem dims ----------------
#define BATCH  8
#define SEQ    2048
#define NMEAS  37
#define HID    256
#define NSTAT  8
#define TEMB   32
#define NL     4
#define DIN    512
#define NST    16
#define RANK   16
#define KCONV  4
#define NLAB   2
#define MROWS  (BATCH*SEQ)   // 16384

typedef __bf16 bf16;
typedef __attribute__((ext_vector_type(8)))  bf16  v8bf;
typedef __attribute__((ext_vector_type(16))) bf16  v16bf;
typedef __attribute__((ext_vector_type(8)))  float v8f;

__device__ __forceinline__ float sigmoidf_(float x){ return 1.f/(1.f+__expf(-x)); }

// ---------------- f32 -> bf16 convert ----------------
__global__ void k_cvt_bf16(const float* __restrict__ s, bf16* __restrict__ d, int n){
  int i = blockIdx.x*blockDim.x + threadIdx.x;
  if (i < n) d[i] = (bf16)s[i];
}

// ---------------- fused embedding + LayerNorm (one block per (b,t) row) ----------------
__global__ void k_embed_ln(const float* __restrict__ ts, const float* __restrict__ stat,
                           const float* __restrict__ tim,
                           const float* __restrict__ tsW, const float* __restrict__ tsb,
                           const float* __restrict__ t2vw, const float* __restrict__ t2vb,
                           const float* __restrict__ timW, const float* __restrict__ timb,
                           const float* __restrict__ stW,  const float* __restrict__ stb,
                           const float* __restrict__ lng,  const float* __restrict__ lnb,
                           float* __restrict__ hout){
  int row = blockIdx.x; int b = row / SEQ; int t = row % SEQ;
  int h = threadIdx.x;
  __shared__ float tvs[TEMB];
  __shared__ float tsv[NMEAS];
  __shared__ float stv[NSTAT];
  __shared__ float red[HID];
  if (h < TEMB){
    float v = tim[b*SEQ + t]*t2vw[h] + t2vb[h];
    tvs[h] = (h==0) ? v : sinf(v);
  }
  if (h < NMEAS) tsv[h] = ts[((size_t)b*NMEAS + h)*SEQ + t];
  if (h < NSTAT) stv[h] = stat[b*NSTAT + h];
  __syncthreads();
  float acc = tsb[h] + timb[h] + stb[h];
  #pragma unroll 1
  for (int m=0; m<NMEAS; ++m) acc += tsv[m]*tsW[h*NMEAS + m];
  #pragma unroll 1
  for (int e=0; e<TEMB; ++e)  acc += tvs[e]*timW[h*TEMB + e];
  #pragma unroll
  for (int s=0; s<NSTAT; ++s) acc += stv[s]*stW[h*NSTAT + s];
  // LayerNorm over HID
  red[h] = acc; __syncthreads();
  for (int s=HID/2; s>0; s>>=1){ if (h<s) red[h]+=red[h+s]; __syncthreads(); }
  float mu = red[0]*(1.f/HID); __syncthreads();
  float c = acc - mu;
  red[h] = c*c; __syncthreads();
  for (int s=HID/2; s>0; s>>=1){ if (h<s) red[h]+=red[h+s]; __syncthreads(); }
  float var = red[0]*(1.f/HID);
  hout[(size_t)row*HID + h] = c*rsqrtf(var + 1e-12f)*lng[h] + lnb[h];
}

// ---------------- rmsnorm (HID row) -> bf16 ----------------
__global__ void k_rmsnorm_bf16(const float* __restrict__ x, const float* __restrict__ w,
                               bf16* __restrict__ out){
  int row = blockIdx.x, h = threadIdx.x;
  __shared__ float red[HID];
  float v = x[(size_t)row*HID + h];
  red[h] = v*v; __syncthreads();
  for (int s=HID/2; s>0; s>>=1){ if (h<s) red[h]+=red[h+s]; __syncthreads(); }
  float rinv = rsqrtf(red[0]*(1.f/HID) + 1e-5f);
  out[(size_t)row*HID + h] = (bf16)(v*rinv*w[h]);
}

// ---------------- WMMA bf16 GEMM: C[m,n] = sum_k A[m,k]*B[n,k] (+res) ----------------
// A: [M,K] row-major bf16 ; B: [N,K] row-major bf16 ; C: [M,N] f32.
// One wave computes a 16x16 tile; fragment loads map to ISA 7.12.2 layouts.
__global__ void k_wmma_gemm(const bf16* __restrict__ A, const bf16* __restrict__ B,
                            float* __restrict__ C, const float* __restrict__ res,
                            int M, int N, int K){
  int wave   = blockIdx.x*(blockDim.x>>5) + (threadIdx.x>>5);
  int lane   = threadIdx.x & 31;
  int ntiles = N >> 4;
  int tiles  = (M>>4)*ntiles;
  if (wave >= tiles) return;
  int tM = (wave / ntiles) << 4;
  int tN = (wave % ntiles) << 4;
  int hi  = lane >> 4;     // half-wave select
  int l16 = lane & 15;

  const bf16* arow = A + (size_t)(tM + l16)*K;   // A row for this lane
  const bf16* brow = B + (size_t)(tN + l16)*K;   // weight row (= B^T column), K-contiguous

  union Frag { v16bf v; v8bf h2[2]; };
  v8f acc = {};
  for (int k0 = 0; k0 < K; k0 += 32){
    Frag fa, fb;
    // A 16x32: lane<16 holds K {k0..k0+7, k0+16..k0+23}; lane>=16 shifts by 8
    fa.h2[0] = *(const v8bf*)(arow + k0 + hi*8);
    fa.h2[1] = *(const v8bf*)(arow + k0 + 16 + hi*8);
    // B 32x16: same K order as A: lane<16 holds K {k0..k0+7, k0+16..k0+23}; lane>=16 shifts by 8
    const bf16* bp = brow + k0 + hi*8;
    fb.h2[0] = *(const v8bf*)(bp);
    fb.h2[1] = *(const v8bf*)(bp + 16);
    if (k0 + 32 < K){
      __builtin_prefetch(arow + k0 + 32, 0, 0);   // global_prefetch_b8
      __builtin_prefetch(brow + k0 + 32, 0, 0);
    }
    acc = __builtin_amdgcn_wmma_f32_16x16x32_bf16(false, fa.v, false, fb.v,
                                                  (short)0, acc, false, false);
  }
  // C layout: VGPR r -> M = r (lanes 0-15) / r+8 (lanes 16-31); N = lane&15
  size_t cbase = (size_t)(tM + hi*8)*N + tN + l16;
  #pragma unroll
  for (int r = 0; r < 8; ++r){
    float v = acc[r];
    if (res) v += res[cbase + (size_t)r*N];
    C[cbase + (size_t)r*N] = v;
  }
}

// ---------------- causal depthwise conv (K=4) + SiLU ----------------
__global__ void k_conv_silu(const float* __restrict__ proj, const float* __restrict__ cW,
                            const float* __restrict__ cb,
                            float* __restrict__ xc, bf16* __restrict__ xcbf){
  int i = blockIdx.x*blockDim.x + threadIdx.x;
  if (i >= MROWS*DIN) return;
  int d = i & (DIN-1);
  int m = i >> 9;           // DIN = 512
  int t = m & (SEQ-1);
  int b = m >> 11;          // SEQ = 2048
  float acc = cb[d];
  #pragma unroll
  for (int k = 0; k < KCONV; ++k){
    int tt = t - (KCONV-1) + k;
    if (tt >= 0)
      acc += cW[d*KCONV + k] * proj[((size_t)(b*SEQ + tt))*(2*DIN) + d];
  }
  float s = acc * sigmoidf_(acc);
  xc[i]   = s;
  xcbf[i] = (bf16)s;
}

// ---------------- dt = softplus(xdbc[:, :16] @ dtW.T + b)  (K=16: VALU, keeps f32) ----------------
__global__ void k_dt(const float* __restrict__ xdbc, const float* __restrict__ dtW,
                     const float* __restrict__ dtb, float* __restrict__ dt){
  int i = blockIdx.x*blockDim.x + threadIdx.x;
  if (i >= MROWS*DIN) return;
  int d = i & (DIN-1);
  int m = i >> 9;
  const float* xr = xdbc + (size_t)m*48;
  float acc = dtb[d];
  #pragma unroll
  for (int r = 0; r < RANK; ++r) acc += xr[r]*dtW[d*RANK + r];
  dt[i] = (acc > 20.f) ? acc : log1pf(__expf(acc));
}

// ---------------- selective scan: lane = (d,n) state; 16-lane butterfly over NST ----------------
// y is aliased onto the dt buffer: dt[b,t,d] is consumed by the same 16-lane group
// (wave-lockstep) before y[b,t,d] is stored in the same iteration.
__global__ void k_scan(const float* __restrict__ xc, const float* __restrict__ dtbuf,
                       const float* __restrict__ xdbc, const float* __restrict__ Alog,
                       const float* __restrict__ Dv, float* __restrict__ y){
  int b  = blockIdx.x >> 5;                 // 32 blocks per batch
  int d  = ((blockIdx.x & 31) << 4) + (threadIdx.x >> 4);
  int n  = threadIdx.x & 15;
  float a  = -__expf(Alog[d*NST + n]);
  float Dd = Dv[d];
  float h = 0.f;
  for (int t = 0; t < SEQ; ++t){
    size_t row = (size_t)b*SEQ + t;
    float dtv = dtbuf[row*DIN + d];
    float u   = xc  [row*DIN + d];
    const float* xr = xdbc + row*48;
    float Bv = xr[RANK + n];
    float Cv = xr[RANK + NST + n];
    float dA = __expf(dtv*a);
    h = dA*h + dtv*Bv*u;
    float p = h*Cv;
    p += __shfl_xor(p, 8);
    p += __shfl_xor(p, 4);
    p += __shfl_xor(p, 2);
    p += __shfl_xor(p, 1);
    if (n == 0) y[row*DIN + d] = p + u*Dd;
  }
}

// ---------------- gate: yg = y * silu(z) -> bf16 ----------------
__global__ void k_gate(const float* __restrict__ y, const float* __restrict__ proj,
                       bf16* __restrict__ ygbf){
  int i = blockIdx.x*blockDim.x + threadIdx.x;
  if (i >= MROWS*DIN) return;
  int d = i & (DIN-1);
  int m = i >> 9;
  float z = proj[(size_t)m*(2*DIN) + DIN + d];
  ygbf[i] = (bf16)(y[i] * z * sigmoidf_(z));
}

// ---------------- final rms per row -> rinv ----------------
__global__ void k_rinv(const float* __restrict__ h, float* __restrict__ rinv){
  int row = blockIdx.x, t = threadIdx.x;
  __shared__ float red[HID];
  float v = h[(size_t)row*HID + t];
  red[t] = v*v; __syncthreads();
  for (int s=HID/2; s>0; s>>=1){ if (t<s) red[t]+=red[t+s]; __syncthreads(); }
  if (t == 0) rinv[row] = rsqrtf(red[0]*(1.f/HID) + 1e-5f);
}

// ---------------- deterministic mean-pool of rmsnormed h ----------------
__global__ void k_pool(const float* __restrict__ h, const float* __restrict__ rinv,
                       const float* __restrict__ normf, float* __restrict__ pooled){
  int i = blockIdx.x*blockDim.x + threadIdx.x;
  if (i >= BATCH*HID) return;
  int b = i >> 8, hh = i & (HID-1);
  float acc = 0.f;
  for (int t = 0; t < SEQ; ++t){
    size_t r = (size_t)b*SEQ + t;
    acc += h[r*HID + hh]*rinv[r];
  }
  pooled[i] = acc*normf[hh]*(1.f/SEQ);
}

// ---------------- dense + relu ----------------
__global__ void k_dense(const float* __restrict__ p, const float* __restrict__ W,
                        const float* __restrict__ bvec, float* __restrict__ z){
  int i = blockIdx.x*blockDim.x + threadIdx.x;
  if (i >= BATCH*HID) return;
  int b = i >> 8, j = i & (HID-1);
  float acc = bvec[j];
  for (int k = 0; k < HID; ++k) acc += p[b*HID + k]*W[j*HID + k];
  z[i] = fmaxf(acc, 0.f);
}

// ---------------- head ----------------
__global__ void k_head(const float* __restrict__ z, const float* __restrict__ W,
                       const float* __restrict__ bvec, float* __restrict__ out){
  int i = blockIdx.x*blockDim.x + threadIdx.x;
  if (i >= BATCH*NLAB) return;
  int b = i / NLAB, l = i % NLAB;
  float acc = bvec[l];
  for (int k = 0; k < HID; ++k) acc += z[b*HID + k]*W[l*HID + k];
  out[i] = acc;
}

// =======================================================================
extern "C" void kernel_launch(void* const* d_in, const int* in_sizes, int n_in,
                              void* d_out, int out_size, void* d_ws, size_t ws_size,
                              hipStream_t stream) {
  const float* ts      = (const float*)d_in[0];
  const float* stat    = (const float*)d_in[1];
  const float* tim     = (const float*)d_in[2];
  const float* tsW     = (const float*)d_in[3];
  const float* tsb     = (const float*)d_in[4];
  const float* t2vw    = (const float*)d_in[5];
  const float* t2vb    = (const float*)d_in[6];
  const float* timW    = (const float*)d_in[7];
  const float* timb    = (const float*)d_in[8];
  const float* stW     = (const float*)d_in[9];
  const float* stb     = (const float*)d_in[10];
  const float* lng     = (const float*)d_in[11];
  const float* lnb     = (const float*)d_in[12];
  const float* norm_w  = (const float*)d_in[13];
  const float* ipW     = (const float*)d_in[14];
  const float* cW      = (const float*)d_in[15];
  const float* cb      = (const float*)d_in[16];
  const float* xpW     = (const float*)d_in[17];
  const float* dtW     = (const float*)d_in[18];
  const float* dtb     = (const float*)d_in[19];
  const float* Alog    = (const float*)d_in[20];
  const float* Dsk     = (const float*)d_in[21];
  const float* opW     = (const float*)d_in[22];
  const float* normf   = (const float*)d_in[23];
  const float* denW    = (const float*)d_in[24];
  const float* denb    = (const float*)d_in[25];
  const float* headW   = (const float*)d_in[26];
  const float* headb   = (const float*)d_in[27];
  float* out = (float*)d_out;

  // ---- workspace carve-out ----
  char* base = (char*)d_ws;
  size_t off = 0;
  auto alloc = [&](size_t bytes)->char*{
    char* p = base + off;
    off = (off + bytes + 255) & ~(size_t)255;
    return p;
  };
  bf16*  wbf_ip = (bf16*) alloc((size_t)NL*1024*HID*sizeof(bf16));
  bf16*  wbf_xp = (bf16*) alloc((size_t)NL*48*DIN*sizeof(bf16));
  bf16*  wbf_op = (bf16*) alloc((size_t)NL*HID*DIN*sizeof(bf16));
  float* hbuf   = (float*)alloc((size_t)MROWS*HID*sizeof(float));
  bf16*  xn     = (bf16*) alloc((size_t)MROWS*HID*sizeof(bf16));
  float* proj   = (float*)alloc((size_t)MROWS*2*DIN*sizeof(float));
  float* xc     = (float*)alloc((size_t)MROWS*DIN*sizeof(float));
  bf16*  xcbf   = (bf16*) alloc((size_t)MROWS*DIN*sizeof(bf16));
  float* xdbc   = (float*)alloc((size_t)MROWS*48*sizeof(float));
  float* dtbuf  = (float*)alloc((size_t)MROWS*DIN*sizeof(float)); // also y (aliased)
  bf16*  ygbf   = (bf16*) alloc((size_t)MROWS*DIN*sizeof(bf16));
  float* rinvb  = (float*)alloc((size_t)MROWS*sizeof(float));
  float* pooled = (float*)alloc((size_t)BATCH*HID*sizeof(float));
  float* densez = (float*)alloc((size_t)BATCH*HID*sizeof(float));
  (void)ws_size; (void)n_in; (void)in_sizes; (void)out_size;

  // ---- convert weights to bf16 once per launch ----
  {
    int n0 = NL*1024*HID, n1 = NL*48*DIN, n2 = NL*HID*DIN;
    k_cvt_bf16<<<(n0+255)/256,256,0,stream>>>(ipW, wbf_ip, n0);
    k_cvt_bf16<<<(n1+255)/256,256,0,stream>>>(xpW, wbf_xp, n1);
    k_cvt_bf16<<<(n2+255)/256,256,0,stream>>>(opW, wbf_op, n2);
  }

  // ---- embedding + LayerNorm ----
  k_embed_ln<<<MROWS,HID,0,stream>>>(ts, stat, tim, tsW, tsb, t2vw, t2vb,
                                     timW, timb, stW, stb, lng, lnb, hbuf);

  const int ED = (MROWS*DIN + 255)/256;  // 32768 blocks for elementwise DIN kernels
  for (int l = 0; l < NL; ++l){
    // rmsnorm -> bf16
    k_rmsnorm_bf16<<<MROWS,HID,0,stream>>>(hbuf, norm_w + l*HID, xn);
    // in_proj: [16384,256] x [1024,256]^T -> proj [16384,1024]
    {
      int tiles = (MROWS/16)*(1024/16);
      k_wmma_gemm<<<(tiles+7)/8,256,0,stream>>>(xn, wbf_ip + (size_t)l*1024*HID,
                                                proj, nullptr, MROWS, 1024, HID);
    }
    // causal conv + silu
    k_conv_silu<<<ED,256,0,stream>>>(proj, cW + l*DIN*KCONV, cb + l*DIN, xc, xcbf);
    // x_proj: [16384,512] x [48,512]^T -> xdbc [16384,48]
    {
      int tiles = (MROWS/16)*(48/16);
      k_wmma_gemm<<<(tiles+7)/8,256,0,stream>>>(xcbf, wbf_xp + (size_t)l*48*DIN,
                                                xdbc, nullptr, MROWS, 48, DIN);
    }
    // dt = softplus(...)
    k_dt<<<ED,256,0,stream>>>(xdbc, dtW + l*DIN*RANK, dtb + l*DIN, dtbuf);
    // selective scan (y aliased onto dtbuf)
    k_scan<<<BATCH*(DIN/16),256,0,stream>>>(xc, dtbuf, xdbc,
                                            Alog + (size_t)l*DIN*NST, Dsk + l*DIN, dtbuf);
    // gate with silu(z)
    k_gate<<<ED,256,0,stream>>>(dtbuf, proj, ygbf);
    // out_proj + residual into h: [16384,512] x [256,512]^T
    {
      int tiles = (MROWS/16)*(HID/16);
      k_wmma_gemm<<<(tiles+7)/8,256,0,stream>>>(ygbf, wbf_op + (size_t)l*HID*DIN,
                                                hbuf, hbuf, MROWS, HID, DIN);
    }
  }

  // ---- final rmsnorm + mean pool (deterministic) + MLP head ----
  k_rinv <<<MROWS,HID,0,stream>>>(hbuf, rinvb);
  k_pool <<<(BATCH*HID+255)/256,256,0,stream>>>(hbuf, rinvb, normf, pooled);
  k_dense<<<(BATCH*HID+255)/256,256,0,stream>>>(pooled, denW, denb, densez);
  k_head <<<1,64,0,stream>>>(densez, headW, headb, out);
}
